// GATAnomalyModel_55946243998297
// MI455X (gfx1250) — compile-verified, matrix operand order checked by reading, not run on hardware
//
#include <hip/hip_runtime.h>
#include <math.h>

#define NEG_SLOPE 0.2f
#define GAT_EPS   1e-16f

typedef __attribute__((ext_vector_type(16))) _Float16 v16h;
typedef __attribute__((ext_vector_type(8)))  float    v8f;

// ---------------------------------------------------------------------------
// helpers
// ---------------------------------------------------------------------------
__device__ __forceinline__ void atomicMaxF(float* addr, float v) {
  // two-sided int trick: works for mixed-sign floats, init must be -inf
  if (v >= 0.f) atomicMax((int*)addr, __float_as_int(v));
  else          atomicMin((unsigned int*)addr, __float_as_uint(v));
}

__global__ void k_fill(float* __restrict__ p, float v, long long n) {
  long long i = (long long)blockIdx.x * blockDim.x + threadIdx.x;
  if (i < n) p[i] = v;
}

// int64 [2,E] edge_index -> int32 src/dst (halves streamed bytes on 6 edge passes)
__global__ void k_cvt_idx(const long long* __restrict__ ei, int* __restrict__ src,
                          int* __restrict__ dst, long long E) {
  long long i = (long long)blockIdx.x * blockDim.x + threadIdx.x;
  if (i < E) { src[i] = (int)ei[i]; dst[i] = (int)ei[E + i]; }
}

// ---------------------------------------------------------------------------
// conv1 node transform: xw1 = x @ W1  (Fin=4 -> VALU; memory bound)
// also alpha_src/alpha_dst per (node, head)
// ---------------------------------------------------------------------------
__global__ void k_conv1_node(const float* __restrict__ x, const float* __restrict__ W1,
                             const float* __restrict__ aS, const float* __restrict__ aD,
                             float* __restrict__ xw, float* __restrict__ as1,
                             float* __restrict__ ad1, int N) {
  int t = blockIdx.x * blockDim.x + threadIdx.x;
  int i = t >> 2, h = t & 3;                       // (node, head)
  if (i >= N) return;
  float x0 = x[i*4+0], x1 = x[i*4+1], x2 = x[i*4+2], x3 = x[i*4+3];
  const float* w = W1 + h * 32;                    // W1 is [4,128] row-major
  float ds = 0.f, dd = 0.f;
  #pragma unroll
  for (int j = 0; j < 32; ++j) {
    float v = x0*w[j] + x1*w[128+j] + x2*w[256+j] + x3*w[384+j];
    xw[(long long)i*128 + h*32 + j] = v;
    ds += v * aS[h*32 + j];
    dd += v * aD[h*32 + j];
  }
  as1[i*4 + h] = ds;
  ad1[i*4 + h] = dd;
}

// ---------------------------------------------------------------------------
// edge softmax pass 1: segment max via float atomics (m init = -inf)
// ---------------------------------------------------------------------------
__global__ void k_edge_max(const int* __restrict__ src32, const int* __restrict__ dst32,
                           const float* __restrict__ as_, const float* __restrict__ ad_,
                           float* __restrict__ m, int H, long long E, int N) {
  long long t = (long long)blockIdx.x * blockDim.x + threadIdx.x;
  long long ET = (E + N) * H;
  if (t >= ET) return;
  long long e = t / H; int h = (int)(t % H);
  int s, d;
  if (e < E) { s = src32[e]; d = dst32[e]; } else { s = d = (int)(e - E); }
  float v = as_[(long long)s*H + h] + ad_[(long long)d*H + h];
  v = v > 0.f ? v : NEG_SLOPE * v;
  atomicMaxF(&m[(long long)d*H + h], v);
}

// ---------------------------------------------------------------------------
// edge softmax pass 2: segment sum of exp(e - m[dst])
// ---------------------------------------------------------------------------
__global__ void k_edge_sum(const int* __restrict__ src32, const int* __restrict__ dst32,
                           const float* __restrict__ as_, const float* __restrict__ ad_,
                           const float* __restrict__ m, float* __restrict__ ssum,
                           int H, long long E, int N) {
  long long t = (long long)blockIdx.x * blockDim.x + threadIdx.x;
  long long ET = (E + N) * H;
  if (t >= ET) return;
  long long e = t / H; int h = (int)(t % H);
  int s, d;
  if (e < E) { s = src32[e]; d = dst32[e]; } else { s = d = (int)(e - E); }
  float v = as_[(long long)s*H + h] + ad_[(long long)d*H + h];
  v = v > 0.f ? v : NEG_SLOPE * v;
  float mv = m[(long long)d*H + h];
  if (!(mv > -3.0e38f)) mv = 0.f;                 // where(isfinite(m), m, 0)
  atomicAdd(&ssum[(long long)d*H + h], expf(v - mv));
}

// ---------------------------------------------------------------------------
// edge softmax pass 3: alpha = ex/(s+eps); scatter msg = feat[src]*alpha into out[dst]
// optional attn output (conv2, H==1)
// ---------------------------------------------------------------------------
__global__ void k_edge_scatter(const int* __restrict__ src32, const int* __restrict__ dst32,
                               const float* __restrict__ as_, const float* __restrict__ ad_,
                               const float* __restrict__ m, const float* __restrict__ ssum,
                               const float* __restrict__ feat, float* __restrict__ out,
                               float* __restrict__ attn, int H, long long E, int N) {
  long long t = (long long)blockIdx.x * blockDim.x + threadIdx.x;
  long long ET = (E + N) * H;
  if (t >= ET) return;
  long long e = t / H; int h = (int)(t % H);
  int s, d;
  if (e < E) { s = src32[e]; d = dst32[e]; } else { s = d = (int)(e - E); }
  float v = as_[(long long)s*H + h] + ad_[(long long)d*H + h];
  v = v > 0.f ? v : NEG_SLOPE * v;
  float mv = m[(long long)d*H + h];
  if (!(mv > -3.0e38f)) mv = 0.f;
  float ex = expf(v - mv);
  float alpha = ex / (ssum[(long long)d*H + h] + GAT_EPS);
  if (attn) attn[t] = alpha;                       // t == e*H + h
  const float* f = feat + ((long long)s*H + h) * 32;
  float*       o = out  + ((long long)d*H + h) * 32;
  #pragma unroll
  for (int j = 0; j < 32; ++j) atomicAdd(&o[j], f[j] * alpha);
}

// ---------------------------------------------------------------------------
// out = elu(out + bias) in place
// ---------------------------------------------------------------------------
__global__ void k_elu_bias(float* __restrict__ v, const float* __restrict__ b,
                           long long n, int F) {
  long long i = (long long)blockIdx.x * blockDim.x + threadIdx.x;
  if (i >= n) return;
  float val = v[i] + b[(int)(i % F)];
  v[i] = val > 0.f ? val : expm1f(val);
}

// ---------------------------------------------------------------------------
// conv2 GEMM: xw2[N,32] = h[N,128] @ W2[128,32] via v_wmma_f32_16x16x32_f16
// f32 -> f16 conversion in-register; f32 accumulate. One wave = 16-node tile.
// A layout (ISA 16-bit 16x32): lane L holds row M=L%16; sel=L/16 picks K runs
//   a[0..7]  = K in [kc*32 + sel*8,      +8)
//   a[8..15] = K in [kc*32 + 16 + sel*8, +8)
// B mirrors with N=L%16 columns. C/D: c[v] -> (M = v + 8*sel, N = L%16).
// ---------------------------------------------------------------------------
__global__ void k_conv2_gemm(const float* __restrict__ hsrc, const float* __restrict__ W2,
                             float* __restrict__ xw2, int N) {
  const int lane = threadIdx.x & 31;
  const int col  = lane & 15;
  const int sel  = lane >> 4;
  const int wavesPerBlock = blockDim.x >> 5;
  const int wave   = blockIdx.x * wavesPerBlock + (threadIdx.x >> 5);
  const int nwaves = gridDim.x * wavesPerBlock;

  // Preload B = W2 (128x32) once per wave: b[n-half][k-chunk], 64 VGPRs
  v16h b[2][4];
  #pragma unroll
  for (int half = 0; half < 2; ++half)
    #pragma unroll
    for (int kc = 0; kc < 4; ++kc)
      #pragma unroll
      for (int j = 0; j < 8; ++j) {
        b[half][kc][j]     = (_Float16)W2[(kc*32 +      sel*8 + j) * 32 + half*16 + col];
        b[half][kc][8 + j] = (_Float16)W2[(kc*32 + 16 + sel*8 + j) * 32 + half*16 + col];
      }

  const int ntiles = (N + 15) >> 4;
  for (int tile = wave; tile < ntiles; tile += nwaves) {
    int row = tile*16 + col;
    if (row >= N) row = N - 1;                     // clamp loads; EXEC stays all-1 for WMMA
    const float* hr = hsrc + (long long)row * 128;
    v16h a[4];
    #pragma unroll
    for (int kc = 0; kc < 4; ++kc)
      #pragma unroll
      for (int j = 0; j < 8; ++j) {
        a[kc][j]     = (_Float16)hr[kc*32 +      sel*8 + j];
        a[kc][8 + j] = (_Float16)hr[kc*32 + 16 + sel*8 + j];
      }
    v8f c0 = {}; v8f c1 = {};
    #pragma unroll
    for (int kc = 0; kc < 4; ++kc) {
      c0 = __builtin_amdgcn_wmma_f32_16x16x32_f16(false, a[kc], false, b[0][kc],
                                                  (short)0, c0, false, false);
      c1 = __builtin_amdgcn_wmma_f32_16x16x32_f16(false, a[kc], false, b[1][kc],
                                                  (short)0, c1, false, false);
    }
    #pragma unroll
    for (int v = 0; v < 8; ++v) {
      int M = v + 8*sel;
      long long r = (long long)tile*16 + M;
      if (r < N) {
        xw2[r*32 + col]      = c0[v];
        xw2[r*32 + 16 + col] = c1[v];
      }
    }
  }
}

// per-node attention logits for conv2 (H=1)
__global__ void k_conv2_alpha(const float* __restrict__ xw2, const float* __restrict__ aS,
                              const float* __restrict__ aD, float* __restrict__ as2,
                              float* __restrict__ ad2, int N) {
  int i = blockIdx.x * blockDim.x + threadIdx.x;
  if (i >= N) return;
  float ds = 0.f, dd = 0.f;
  #pragma unroll
  for (int f = 0; f < 32; ++f) {
    float v = xw2[(long long)i*32 + f];
    ds += v * aS[f];
    dd += v * aD[f];
  }
  as2[i] = ds; ad2[i] = dd;
}

// ---------------------------------------------------------------------------
// mean pool over nodes (32 features) -> pool[32] (pre-divided at logits)
// ---------------------------------------------------------------------------
__global__ void k_pool(const float* __restrict__ out2, float* __restrict__ pool, int N) {
  __shared__ float sh[256];
  int f = threadIdx.x & 31;
  int g = threadIdx.x >> 5;                        // 8 node-subgroups per block
  float acc = 0.f;
  for (long long i = (long long)blockIdx.x * 8 + g; i < N; i += (long long)gridDim.x * 8)
    acc += out2[i*32 + f];
  sh[threadIdx.x] = acc;
  __syncthreads();
  if (g == 0) {
    float a = acc;
    #pragma unroll
    for (int k = 1; k < 8; ++k) a += sh[k*32 + f];
    atomicAdd(&pool[f], a);
  }
}

__global__ void k_logits(const float* __restrict__ pool, const float* __restrict__ lin_w,
                         const float* __restrict__ lin_b, float* __restrict__ out, int N) {
  int j = threadIdx.x;
  if (j >= 2) return;
  float acc = 0.f;
  float inv = 1.0f / (float)N;
  #pragma unroll
  for (int f = 0; f < 32; ++f) acc += (pool[f] * inv) * lin_w[f*2 + j];
  out[j] = acc + lin_b[j];
}

// ---------------------------------------------------------------------------
// launch
// ---------------------------------------------------------------------------
extern "C" void kernel_launch(void* const* d_in, const int* in_sizes, int n_in,
                              void* d_out, int out_size, void* d_ws, size_t ws_size,
                              hipStream_t stream) {
  (void)n_in; (void)out_size; (void)ws_size;
  const float*     x    = (const float*)d_in[0];
  const long long* ei   = (const long long*)d_in[1];
  const float*     W1   = (const float*)d_in[2];
  const float*     aS1  = (const float*)d_in[3];
  const float*     aD1  = (const float*)d_in[4];
  const float*     b1   = (const float*)d_in[5];
  const float*     W2   = (const float*)d_in[6];
  const float*     aS2  = (const float*)d_in[7];
  const float*     aD2  = (const float*)d_in[8];
  const float*     b2   = (const float*)d_in[9];
  const float*     linw = (const float*)d_in[10];
  const float*     linb = (const float*)d_in[11];

  const int       N  = in_sizes[0] / 4;
  const long long E  = (long long)in_sizes[1] / 2;
  const long long ET = E + N;

  float* out   = (float*)d_out;     // [0..1] logits, [2 .. 2+ET) attn (conv2 alpha)
  float* attn  = out + 2;

  // workspace carve (256B aligned)
  char* wp = (char*)d_ws;
  auto alloc = [&](size_t bytes) -> char* {
    char* p = wp; wp += (bytes + 255) & ~(size_t)255; return p;
  };
  int*   src32 = (int*)  alloc((size_t)E * 4);
  int*   dst32 = (int*)  alloc((size_t)E * 4);
  float* xw1   = (float*)alloc((size_t)N * 128 * 4);
  float* as1   = (float*)alloc((size_t)N * 4 * 4);
  float* ad1   = (float*)alloc((size_t)N * 4 * 4);
  float* m1    = (float*)alloc((size_t)N * 4 * 4);
  float* s1    = (float*)alloc((size_t)N * 4 * 4);
  float* out1  = (float*)alloc((size_t)N * 128 * 4);   // becomes h after elu+bias
  float* xw2   = (float*)alloc((size_t)N * 32 * 4);
  float* as2   = (float*)alloc((size_t)N * 4);
  float* ad2   = (float*)alloc((size_t)N * 4);
  float* m2    = (float*)alloc((size_t)N * 4);
  float* s2    = (float*)alloc((size_t)N * 4);
  float* out2  = (float*)alloc((size_t)N * 32 * 4);
  float* pool  = (float*)alloc(32 * 4);

  const int B = 256;
  auto nb = [](long long n, int b) -> unsigned { return (unsigned)((n + b - 1) / b); };
  const float NEG_INF = -__builtin_huge_valf();

  // index compaction
  hipLaunchKernelGGL(k_cvt_idx, dim3(nb(E, B)), dim3(B), 0, stream, ei, src32, dst32, E);

  // ---- conv1 ----
  hipLaunchKernelGGL(k_conv1_node, dim3(nb((long long)N * 4, B)), dim3(B), 0, stream,
                     x, W1, aS1, aD1, xw1, as1, ad1, N);
  hipLaunchKernelGGL(k_fill, dim3(nb((long long)N * 4, B)), dim3(B), 0, stream, m1, NEG_INF, (long long)N * 4);
  hipLaunchKernelGGL(k_fill, dim3(nb((long long)N * 4, B)), dim3(B), 0, stream, s1, 0.f, (long long)N * 4);
  hipLaunchKernelGGL(k_fill, dim3(nb((long long)N * 128, B)), dim3(B), 0, stream, out1, 0.f, (long long)N * 128);
  hipLaunchKernelGGL(k_edge_max,     dim3(nb(ET * 4, B)), dim3(B), 0, stream,
                     src32, dst32, as1, ad1, m1, 4, E, N);
  hipLaunchKernelGGL(k_edge_sum,     dim3(nb(ET * 4, B)), dim3(B), 0, stream,
                     src32, dst32, as1, ad1, m1, s1, 4, E, N);
  hipLaunchKernelGGL(k_edge_scatter, dim3(nb(ET * 4, B)), dim3(B), 0, stream,
                     src32, dst32, as1, ad1, m1, s1, xw1, out1, (float*)nullptr, 4, E, N);
  hipLaunchKernelGGL(k_elu_bias, dim3(nb((long long)N * 128, B)), dim3(B), 0, stream,
                     out1, b1, (long long)N * 128, 128);

  // ---- conv2 ----
  hipLaunchKernelGGL(k_conv2_gemm, dim3(256), dim3(128), 0, stream, out1, W2, xw2, N);
  hipLaunchKernelGGL(k_conv2_alpha, dim3(nb(N, B)), dim3(B), 0, stream, xw2, aS2, aD2, as2, ad2, N);
  hipLaunchKernelGGL(k_fill, dim3(nb(N, B)), dim3(B), 0, stream, m2, NEG_INF, (long long)N);
  hipLaunchKernelGGL(k_fill, dim3(nb(N, B)), dim3(B), 0, stream, s2, 0.f, (long long)N);
  hipLaunchKernelGGL(k_fill, dim3(nb((long long)N * 32, B)), dim3(B), 0, stream, out2, 0.f, (long long)N * 32);
  hipLaunchKernelGGL(k_fill, dim3(1), dim3(32), 0, stream, pool, 0.f, (long long)32);
  hipLaunchKernelGGL(k_edge_max,     dim3(nb(ET, B)), dim3(B), 0, stream,
                     src32, dst32, as2, ad2, m2, 1, E, N);
  hipLaunchKernelGGL(k_edge_sum,     dim3(nb(ET, B)), dim3(B), 0, stream,
                     src32, dst32, as2, ad2, m2, s2, 1, E, N);
  hipLaunchKernelGGL(k_edge_scatter, dim3(nb(ET, B)), dim3(B), 0, stream,
                     src32, dst32, as2, ad2, m2, s2, xw2, out2, attn, 1, E, N);
  hipLaunchKernelGGL(k_elu_bias, dim3(nb((long long)N * 32, B)), dim3(B), 0, stream,
                     out2, b2, (long long)N * 32, 32);

  // ---- pool + linear ----
  hipLaunchKernelGGL(k_pool, dim3(256), dim3(256), 0, stream, out2, pool, N);
  hipLaunchKernelGGL(k_logits, dim3(1), dim3(32), 0, stream, pool, linw, linb, out, N);
}
